// GraphEncoder_16956530884765
// MI455X (gfx1250) — compile-verified, matrix operand order checked by reading, not run on hardware
//
#include <hip/hip_runtime.h>
#include <hip/hip_bf16.h>
#include <math.h>

// ---------------------------------------------------------------------------
// GCN-style 2-layer graph encoder for MI455X (gfx1250, wave32, WMMA).
//   deg -> norm = rsqrt(max(deg,1))
//   h0 = features*norm ; agg1 = scatter(h0[src]*ew -> dst)
//   h1 = tanh(agg1@W1+b1)*norm ; agg2 = scatter(h1[src]*ew -> dst)
//   t  = tanh(agg2@W2+b2) ; out = mean(t, axis=0)
// GEMMs use v_wmma_f32_16x16x32_bf16 (bf16 operands, f32 accumulate).
// f32->bf16 via native fptrunc (hardware cvt when the backend has it),
// tanh via v_tanh_f32 (TRANS op, co-executes with WMMA).
// ---------------------------------------------------------------------------

typedef __attribute__((ext_vector_type(16))) __bf16 v16bf;
typedef __attribute__((ext_vector_type(2)))  __bf16 v2bf;
typedef __attribute__((ext_vector_type(8)))  float  v8f;

#define D 128          // feature dim (D_IN == D_HID == 128)
#define TILES_K 4      // 128 / 32
#define TILES_N 8      // 128 / 16
#define TILE_ELEMS 512 // 32x16 bf16 elements per packed B tile

// Native fptrunc: lowers to v_cvt_pk_bf16_f32 / hardware cvt on gfx1250 if
// available, else LLVM's RNE software expansion (same cost as manual trick).
__device__ __forceinline__ __bf16 f2bf(float f) {
#if __has_builtin(__builtin_amdgcn_cvt_pk_bf16_f32)
    v2bf p = __builtin_amdgcn_cvt_pk_bf16_f32(f, 0.0f);
    return p[0];
#else
    return (__bf16)f;
#endif
}

__device__ __forceinline__ v2bf f2bf_pk(float a, float b) {
#if __has_builtin(__builtin_amdgcn_cvt_pk_bf16_f32)
    return __builtin_amdgcn_cvt_pk_bf16_f32(a, b);
#else
    v2bf r; r[0] = (__bf16)a; r[1] = (__bf16)b; return r;
#endif
}

// Hardware tanh (v_tanh_f32) when available, else OCML tanhf.
__device__ __forceinline__ float fast_tanh(float x) {
#if __has_builtin(__builtin_amdgcn_tanhf)
    return __builtin_amdgcn_tanhf(x);
#elif __has_builtin(__builtin_amdgcn_tanh_f32)
    return __builtin_amdgcn_tanh_f32(x);
#else
    return tanhf(x);
#endif
}

// ---- degree: deg[dst[e]] += 1 ------------------------------------------------
__global__ void k_degree(const int* __restrict__ dst, float* __restrict__ deg, int nE) {
    int i = blockIdx.x * blockDim.x + threadIdx.x;
    if (i < nE) unsafeAtomicAdd(&deg[dst[i]], 1.0f);
}

// ---- norm = rsqrt(max(deg,1)) in place --------------------------------------
__global__ void k_norm(float* __restrict__ nrm, int n) {
    int i = blockIdx.x * blockDim.x + threadIdx.x;
    if (i < n) nrm[i] = rsqrtf(fmaxf(nrm[i], 1.0f));
}

// ---- h0 = features * norm[row] (float4 per thread) --------------------------
__global__ void k_scale(const float* __restrict__ f, const float* __restrict__ nrm,
                        float* __restrict__ out, int nQuads) {
    int i = blockIdx.x * blockDim.x + threadIdx.x;
    if (i >= nQuads) return;
    int row = i >> 5;                       // 32 float4 per 128-wide row
    float s = nrm[row];
    float4 v = ((const float4*)f)[i];
    v.x *= s; v.y *= s; v.z *= s; v.w *= s;
    ((float4*)out)[i] = v;
}

// ---- pack W (128x128 f32, row-major) into bf16 WMMA B-operand tiles ---------
// Tile (ct, kt) covers cols [ct*16,ct*16+16), K [kt*32,kt*32+32).
// B layout mirrors the documented 16-bit A layout over N:
//   lane<16 : n=lane,    elems 0..7 -> K = kbase+{half*8 ..}, elems 8..15 -> +16
//   lane>=16: n=lane-16, K blocks {8..15} and {24..31}
__global__ void k_packw(const float* __restrict__ W, __bf16* __restrict__ Wp) {
    int t = blockIdx.x * blockDim.x + threadIdx.x;      // (tile, lane)
    if (t >= 32 * 32) return;
    int lane = t & 31, tile = t >> 5;
    int kt = tile & 3, ct = tile >> 2;
    int n = lane & 15, half = lane >> 4;
    #pragma unroll
    for (int e = 0; e < 16; ++e) {
        int kblk = (e < 8) ? (half * 8 + e) : (16 + half * 8 + (e - 8));
        int k = kt * 32 + kblk;
        Wp[(size_t)tile * TILE_ELEMS + lane * 16 + e] = f2bf(W[k * D + ct * 16 + n]);
    }
}

// ---- scatter: one wave per edge; lane handles 4 contiguous floats -----------
__global__ __launch_bounds__(256)
void k_scatter(const float* __restrict__ h, const float* __restrict__ ew,
               const int* __restrict__ src, const int* __restrict__ dst,
               float* __restrict__ out, int nE) {
    int wave = (blockIdx.x * blockDim.x + threadIdx.x) >> 5;
    int lane = threadIdx.x & 31;
    if (wave >= nE) return;
    int s = src[wave];
    int d = dst[wave];
    float w = ew[wave];
    float4 v = ((const float4*)(h + (size_t)s * D))[lane];
    float* op = out + (size_t)d * D + lane * 4;
    unsafeAtomicAdd(op + 0, v.x * w);
    unsafeAtomicAdd(op + 1, v.y * w);
    unsafeAtomicAdd(op + 2, v.z * w);
    unsafeAtomicAdd(op + 3, v.w * w);
}

// ---- WMMA GEMM: out[16 rows x 128] = tanh(A@W + b) (* norm[row] if useNorm) -
// One wave per 16-row strip. 4 k-steps x 8 col-tiles = 32 v_wmma per wave.
__global__ __launch_bounds__(32)
void k_gemm_tanh(const float* __restrict__ A, const __bf16* __restrict__ Wp,
                 const float* __restrict__ bias, const float* __restrict__ nrm,
                 float* __restrict__ out, int useNorm) {
    const int rowBase = blockIdx.x * 16;
    const int lane = threadIdx.x;
    const int half = lane >> 4;
    const int m = lane & 15;

    v8f acc[TILES_N] = {};

    const float* arow = A + (size_t)(rowBase + m) * D;
    const v16bf* bp = (const v16bf*)Wp;

    #pragma unroll
    for (int kt = 0; kt < TILES_K; ++kt) {
        // A fragment: 16-bit A 16x32 layout (lanes 0-15: K {0..7,16..23}; 16-31: {8..15,24..31})
        int klo = kt * 32 + half * 8;
        int khi = kt * 32 + 16 + half * 8;
        float4 f0 = ((const float4*)(arow + klo))[0];
        float4 f1 = ((const float4*)(arow + klo))[1];
        float4 f2 = ((const float4*)(arow + khi))[0];
        float4 f3 = ((const float4*)(arow + khi))[1];
        v16bf a;
        v2bf p;
        p = f2bf_pk(f0.x, f0.y); a[0]  = p[0]; a[1]  = p[1];
        p = f2bf_pk(f0.z, f0.w); a[2]  = p[0]; a[3]  = p[1];
        p = f2bf_pk(f1.x, f1.y); a[4]  = p[0]; a[5]  = p[1];
        p = f2bf_pk(f1.z, f1.w); a[6]  = p[0]; a[7]  = p[1];
        p = f2bf_pk(f2.x, f2.y); a[8]  = p[0]; a[9]  = p[1];
        p = f2bf_pk(f2.z, f2.w); a[10] = p[0]; a[11] = p[1];
        p = f2bf_pk(f3.x, f3.y); a[12] = p[0]; a[13] = p[1];
        p = f2bf_pk(f3.z, f3.w); a[14] = p[0]; a[15] = p[1];

        #pragma unroll
        for (int ct = 0; ct < TILES_N; ++ct) {
            v16bf b = bp[(size_t)(ct * TILES_K + kt) * 32 + lane];
            acc[ct] = __builtin_amdgcn_wmma_f32_16x16x32_bf16(
                false, a, false, b, (short)0, acc[ct], false, false);
        }
    }

    // C layout: lane<16 -> N=lane, VGPR r -> M=r ; lane>=16 -> N=lane-16, M=8+r
    const int n = lane & 15;
    #pragma unroll
    for (int ct = 0; ct < TILES_N; ++ct) {
        float bv = bias[ct * 16 + n];
        #pragma unroll
        for (int r = 0; r < 8; ++r) {
            int row = rowBase + half * 8 + r;
            float v = fast_tanh(acc[ct][r] + bv);
            if (useNorm) v *= nrm[row];
            out[(size_t)row * D + ct * 16 + n] = v;
        }
    }
}

// ---- column mean: out[c] += sum_r t[r,c] / N --------------------------------
__global__ __launch_bounds__(128)
void k_colmean(const float* __restrict__ t, float* __restrict__ out, int nRows) {
    int c = threadIdx.x;
    float s = 0.0f;
    for (int r = blockIdx.x; r < nRows; r += gridDim.x)
        s += t[(size_t)r * D + c];
    unsafeAtomicAdd(&out[c], s * (1.0f / 100000.0f));
}

// ---------------------------------------------------------------------------
extern "C" void kernel_launch(void* const* d_in, const int* in_sizes, int n_in,
                              void* d_out, int out_size, void* d_ws, size_t ws_size,
                              hipStream_t stream) {
    const float* features = (const float*)d_in[0];
    const float* ew       = (const float*)d_in[1];
    const int*   src      = (const int*)d_in[2];
    const int*   dst      = (const int*)d_in[3];
    const float* W1       = (const float*)d_in[4];
    const float* b1       = (const float*)d_in[5];
    const float* W2       = (const float*)d_in[6];
    const float* b2       = (const float*)d_in[7];
    float* out = (float*)d_out;

    const int N  = in_sizes[0] / D;   // 100000
    const int E  = in_sizes[2];       // 1600000
    const size_t NB = (size_t)N * D * sizeof(float);   // 51.2 MB

    // workspace layout
    char* ws = (char*)d_ws;
    const size_t OFF_NORM = 0;                              // N floats
    const size_t OFF_WP   = 400384;                         // 2 x 32 tiles x 512 bf16
    const size_t OFF_A    = 465920;                         // N*128 f32
    const size_t OFF_B    = OFF_A + 51200000;               // N*128 f32
    float*  nrm  = (float*)(ws + OFF_NORM);
    __bf16* w1p  = (__bf16*)(ws + OFF_WP);
    __bf16* w2p  = w1p + 32 * TILE_ELEMS;
    float*  bufA = (float*)(ws + OFF_A);
    float*  bufB = (float*)(ws + OFF_B);

    // degree -> norm
    hipMemsetAsync(nrm, 0, (size_t)N * sizeof(float), stream);
    hipMemsetAsync(bufB, 0, NB, stream);
    k_degree<<<(E + 255) / 256, 256, 0, stream>>>(dst, nrm, E);
    k_norm<<<(N + 255) / 256, 256, 0, stream>>>(nrm, N);

    // pack weights to bf16 WMMA tiles
    k_packw<<<4, 256, 0, stream>>>(W1, w1p);
    k_packw<<<4, 256, 0, stream>>>(W2, w2p);

    // h0 = features * norm
    int nQuads = N * (D / 4);
    k_scale<<<(nQuads + 255) / 256, 256, 0, stream>>>(features, nrm, bufA, nQuads);

    // agg1 = scatter(h0)
    k_scatter<<<(E * 32 + 255) / 256, 256, 0, stream>>>(bufA, ew, src, dst, bufB, E);

    // h1 = tanh(agg1 @ W1 + b1) * norm
    k_gemm_tanh<<<N / 16, 32, 0, stream>>>(bufB, w1p, b1, nrm, bufA, 1);

    // agg2 = scatter(h1)
    hipMemsetAsync(bufB, 0, NB, stream);
    k_scatter<<<(E * 32 + 255) / 256, 256, 0, stream>>>(bufA, ew, src, dst, bufB, E);

    // t = tanh(agg2 @ W2 + b2)
    k_gemm_tanh<<<N / 16, 32, 0, stream>>>(bufB, w2p, b2, nrm, bufA, 0);

    // out = mean(t, axis=0)
    hipMemsetAsync(out, 0, 128 * sizeof(float), stream);
    k_colmean<<<200, 128, 0, stream>>>(bufA, out, N);
}
